// SelfAttention_16973710754005
// MI455X (gfx1250) — matrix operand order repaired and checked by reading, not compiled, not run on hardware
//
#include <hip/hip_runtime.h>
#include <hip/hip_bf16.h>
#include <cstdint>

// ---------------------------------------------------------------------------
// MI455X (gfx1250) fused self-attention + residual + LayerNorm.
//   B=4, S=2048, D=1024, fp32 in/out.
// Strategy: bf16 hi/lo split (3x v_wmma_f32_16x16x32_bf16) for ~fp32 accuracy
// at ~8x the per-instruction FLOPs of the f32 16x16x4 WMMA. Flash-attention
// style online softmax, V stored transposed for contiguous B-fragments.
// Software-pipelined GEMM staging (global loads overlap WMMA), and TDM
// tensor_load_to_lds (6-arg toolchain variant) for the Q-tile copy.
// ---------------------------------------------------------------------------

typedef __bf16 bf16;
typedef __attribute__((ext_vector_type(8)))  __bf16 v8bf;
typedef __attribute__((ext_vector_type(16))) __bf16 v16bf;
typedef __attribute__((ext_vector_type(8)))  float  v8f;
typedef __attribute__((ext_vector_type(4)))  unsigned int v4u;
typedef __attribute__((ext_vector_type(4)))  int v4i;
typedef __attribute__((ext_vector_type(8)))  int v8i;

#define D_DIM   1024
#define S_SEQ   2048
#define R_TOT   8192      // B*S rows
#define QK_SCALE 0.03125f // 1/sqrt(1024)
#define LN_EPS  1e-3f

#if defined(__has_builtin)
#if __has_builtin(__builtin_amdgcn_tensor_load_to_lds) && \
    __has_builtin(__builtin_amdgcn_s_wait_tensorcnt)
#define USE_TDM 1
#endif
#endif
#ifndef USE_TDM
#define USE_TDM 0
#endif

static __device__ inline v8f vzero8() {
  v8f v;
#pragma unroll
  for (int i = 0; i < 8; ++i) v[i] = 0.0f;
  return v;
}

static __device__ inline v8f wmma_bf16(v16bf a, v16bf b, v8f c) {
  // (neg_a, A, neg_b, B, c_mod, C, reuse_a, reuse_b)
  return __builtin_amdgcn_wmma_f32_16x16x32_bf16(false, a, false, b, (short)0, c,
                                                 false, false);
}

// 16-bit A-matrix fragment (16x32): per lane one contiguous 16-elem chunk at
// k = 2*kbase..2*kbase+15, kbase = (lane>=16) ? 8 : 0 (matches B fragment).
static __device__ inline v16bf load_a_frag(const bf16* base, int kbase) {
  union { v16bf v; v8bf h[2]; } u;
  u.h[0] = *(const v8bf*)(base + 2 * kbase);
  u.h[1] = *(const v8bf*)(base + 2 * kbase + 8);
  return u.v;
}

// 16-bit B-matrix fragment (32x16): per lane one contiguous 16-elem chunk
// (lanes 0-15: k=0..15, lanes 16-31: k=16..31). Caller applies the +16 offset.
static __device__ inline v16bf load_b_frag(const bf16* p) {
  return *(const v16bf*)p;
}

static __device__ inline bf16 to_bf(float f) { return (bf16)f; }

#if USE_TDM
// Issue one TDM DMA: contiguous 2D tile (rows x cols, bf16) global -> LDS.
// D# layout per CDNA5 ISA (group0: count/lds/global/type,
// group1: data_size=2B, tensor dims/strides, tile dims; groups 2/3 unused).
// This toolchain exposes the 6-arg builtin:
//   (uint32x4 g0, int32x8 g1, int32x4 g2, int32x4 g3, int32x8 g4, i32 cpol)
static __device__ inline void tdm_load_2d_bf16(const void* gptr,
                                               unsigned lds_off, int rows,
                                               int cols) {
  unsigned long long ga = (unsigned long long)gptr;
  v4u g0;
  g0[0] = 1u;                                   // count=1, user mode
  g0[1] = lds_off;                              // LDS byte address
  g0[2] = (unsigned)(ga & 0xFFFFFFFFull);       // global addr [31:0]
  g0[3] = (unsigned)((ga >> 32) & 0x01FFFFFFull) | (2u << 30);  // addr hi|type=2
  v8i g1;
  g1[0] = 1 << 16;                              // data_size=1 (2 bytes)
  g1[1] = (cols & 0xFFFF) << 16;                // tensor_dim0 [15:0] at bit48
  g1[2] = ((unsigned)cols >> 16) | ((rows & 0xFFFF) << 16);  // dim0 hi | dim1 lo
  g1[3] = (cols & 0xFFFF) << 16;                // tile_dim0 at bit112
  g1[4] = rows & 0xFFFF;                        // tile_dim1 (tile_dim2=0)
  g1[5] = cols;                                 // tensor_dim0_stride [31:0]
  g1[6] = 0;                                    // stride hi, dim1_stride lo
  g1[7] = 0;
  v4i gz4;
  gz4[0] = 0; gz4[1] = 0; gz4[2] = 0; gz4[3] = 0;
  v8i gz8;
#pragma unroll
  for (int i = 0; i < 8; ++i) gz8[i] = 0;
  __builtin_amdgcn_tensor_load_to_lds(g0, g1, gz4, gz4, gz8, 0);
}
#endif

// ---------------------------------------------------------------------------
// Kernel 1: one of the QKV projections.  out = x[8192,1024] * W[1024,1024] + b
// mode 0: Q  -> hi/lo bf16, pre-scaled by 1/sqrt(D)
// mode 1: K  -> hi/lo bf16
// mode 2: V  -> bf16 (hi only), stored TRANSPOSED: Vt[d][row]
// Block: 256 threads (8 waves). Tile: 128 rows x 128 cols, K-chunks of 32.
// Software-pipelined: chunk kc+1 is fetched into registers while chunk kc's
// WMMAs run; conversion + LDS commit happens at the top of the next iteration.
// ---------------------------------------------------------------------------
__global__ __launch_bounds__(256) void qkv_tile_kernel(
    const float* __restrict__ x, const float* __restrict__ W,
    const float* __restrict__ bias, bf16* __restrict__ outHi,
    bf16* __restrict__ outLo, int mode) {
  __shared__ bf16 sxh[128 * 32];   // x hi, [row][k]
  __shared__ bf16 sxl[128 * 32];   // x lo
  __shared__ bf16 swh[128 * 32];   // W hi, transposed in LDS: [n][k]
  __shared__ bf16 swl[128 * 32];   // W lo

  const int tid  = threadIdx.x;
  const int lane = tid & 31;
  const int wv   = tid >> 5;
  const int ln15 = lane & 15;
  const int hl   = (lane >> 4) & 1;
  const int hl8  = hl * 8;
  const int hl16 = hl * 16;
  const int m0 = blockIdx.x * 128;
  const int n0 = blockIdx.y * 128;

  // staging assignment: 16 contiguous floats of x and of W per thread
  const int xr_r = tid >> 1, xr_c = (tid & 1) * 16;   // x: [row][k-chunk half]
  const int wr_k = tid >> 3, wr_n = (tid & 7) * 16;   // W: [k][n-seg]

  float xr[16], wr[16];
  {  // prefetch chunk 0
    const float* gx = x + (size_t)(m0 + xr_r) * D_DIM + xr_c;
    const float* gw = W + (size_t)wr_k * D_DIM + n0 + wr_n;
#pragma unroll
    for (int j = 0; j < 16; ++j) { xr[j] = gx[j]; wr[j] = gw[j]; }
  }

  v8f acc[8];
#pragma unroll
  for (int nt = 0; nt < 8; ++nt) acc[nt] = vzero8();

  for (int kc = 0; kc < 32; ++kc) {
    // commit prefetched chunk kc to LDS as hi/lo bf16
#pragma unroll
    for (int j = 0; j < 16; ++j) {
      float f = xr[j];
      bf16 h = to_bf(f);
      sxh[xr_r * 32 + xr_c + j] = h;
      sxl[xr_r * 32 + xr_c + j] = to_bf(f - (float)h);
      float fw = wr[j];
      bf16 hw = to_bf(fw);
      swh[(wr_n + j) * 32 + wr_k] = hw;
      swl[(wr_n + j) * 32 + wr_k] = to_bf(fw - (float)hw);
    }
    __syncthreads();

    if (kc + 1 < 32) {  // kick off chunk kc+1 loads; overlap with WMMAs below
      const float* gx = x + (size_t)(m0 + xr_r) * D_DIM + (kc + 1) * 32 + xr_c;
      const float* gw = W + (size_t)((kc + 1) * 32 + wr_k) * D_DIM + n0 + wr_n;
#pragma unroll
      for (int j = 0; j < 16; ++j) { xr[j] = gx[j]; wr[j] = gw[j]; }
    }

    const bf16* arh = sxh + (wv * 16 + ln15) * 32;
    const bf16* arl = sxl + (wv * 16 + ln15) * 32;
    v16bf ah = load_a_frag(arh, hl8);
    v16bf al = load_a_frag(arl, hl8);
#pragma unroll
    for (int nt = 0; nt < 8; ++nt) {
      v16bf bh = load_b_frag(swh + (nt * 16 + ln15) * 32 + hl16);
      v16bf bl = load_b_frag(swl + (nt * 16 + ln15) * 32 + hl16);
      acc[nt] = wmma_bf16(ah, bh, acc[nt]);  // hi*hi
      acc[nt] = wmma_bf16(ah, bl, acc[nt]);  // hi*lo
      acc[nt] = wmma_bf16(al, bh, acc[nt]);  // lo*hi
    }
    __syncthreads();
  }

  // C layout: VGPR i -> row M = i (lanes 0-15) / i+8 (lanes 16-31), col = lane&15
#pragma unroll
  for (int nt = 0; nt < 8; ++nt) {
    const int col = n0 + nt * 16 + ln15;
    const float bv = bias[col];
#pragma unroll
    for (int i = 0; i < 8; ++i) {
      float v = acc[nt][i] + bv;
      if (mode == 0) v *= QK_SCALE;
      const int row = m0 + wv * 16 + i + hl8;
      bf16 h = to_bf(v);
      if (mode == 2) {
        outHi[(size_t)col * R_TOT + row] = h;  // transposed V
      } else {
        outHi[(size_t)row * D_DIM + col] = h;
        outLo[(size_t)row * D_DIM + col] = to_bf(v - (float)h);
      }
    }
  }
}

// ---------------------------------------------------------------------------
// Kernel 2: flash attention + residual + LayerNorm.
// One workgroup (8 waves, 256 threads) per (batch, 16-query tile).
// Wave w: owns 16 keys of each 128-key block for scores, and output columns
// [w*128, w*128+128) for the O = P*V accumulation.
// Dynamic LDS: Q hi/lo tile (64KB) + score/prob buffers + reductions.
// Q tile staged by the Tensor Data Mover when available.
// ---------------------------------------------------------------------------
__global__ __launch_bounds__(256) void attn_kernel(
    const float* __restrict__ x, const bf16* __restrict__ Qhi,
    const bf16* __restrict__ Qlo, const bf16* __restrict__ Khi,
    const bf16* __restrict__ Klo, const bf16* __restrict__ Vt,
    const float* __restrict__ gamma, const float* __restrict__ beta,
    float* __restrict__ out) {
  extern __shared__ char smem[];
  bf16*  qh_s  = (bf16*)smem;                 // 16*1024  (LDS offset 0)
  bf16*  ql_s  = qh_s + 16 * D_DIM;           // 16*1024  (LDS offset 32768)
  float* Sb    = (float*)(ql_s + 16 * D_DIM); // 16*128 scores
  bf16*  Pb    = (bf16*)(Sb + 16 * 128);      // 16*128 probs (bf16)
  float* mrow  = (float*)(Pb + 16 * 128);     // 16 running max
  float* lrow  = mrow + 16;                   // 16 running sum
  float* crow  = lrow + 16;                   // 16 correction factor
  float* red   = crow + 16;                   // 16*16 reduction scratch
  float* sumr  = red + 256;                   // 16 LN sum
  float* sumsq = sumr + 16;                   // 16 LN sum-of-squares

  const int tid  = threadIdx.x;
  const int lane = tid & 31;
  const int wv   = tid >> 5;
  const int ln15 = lane & 15;
  const int hl   = (lane >> 4) & 1;
  const int hl8  = hl * 8;
  const int hl16 = hl * 16;

  const int blk    = blockIdx.x;       // 0..511
  const int qbase  = blk * 16;         // first global row of query tile
  const int batch  = blk >> 7;         // 128 query tiles per batch
  const int kstart = batch * S_SEQ;    // first key row of this batch

  // Stage Q tile (hi+lo) into LDS: 2 x 32KB.
#if USE_TDM
  if (wv == 0) {  // one wave issues the DMAs (TDM ignores EXEC; avoid dup)
    tdm_load_2d_bf16(Qhi + (size_t)qbase * D_DIM, 0u, 16, D_DIM);
    tdm_load_2d_bf16(Qlo + (size_t)qbase * D_DIM, 16u * D_DIM * 2u, 16, D_DIM);
    __builtin_amdgcn_s_wait_tensorcnt((short)0);
  }
#else
  {
    const uint4* srch = (const uint4*)(Qhi + (size_t)qbase * D_DIM);
    const uint4* srcl = (const uint4*)(Qlo + (size_t)qbase * D_DIM);
    uint4* dsth = (uint4*)qh_s;
    uint4* dstl = (uint4*)ql_s;
#pragma unroll
    for (int j = 0; j < 8; ++j) {
      dsth[tid + j * 256] = srch[tid + j * 256];
      dstl[tid + j * 256] = srcl[tid + j * 256];
    }
  }
#endif
  if (tid < 16) { mrow[tid] = -3.0e38f; lrow[tid] = 0.0f; }
  v8f acc[8];
#pragma unroll
  for (int nt = 0; nt < 8; ++nt) acc[nt] = vzero8();
  __syncthreads();

  for (int kb = 0; kb < 16; ++kb) {           // 16 blocks of 128 keys
    const int keyblock = kstart + kb * 128;
    // ---- scores: S[16q x 16k] per wave, K reduced over D in 32-chunks ----
    const int mykey = keyblock + wv * 16 + ln15;  // key row for B-fragment
    const bf16* krh = Khi + (size_t)mykey * D_DIM + hl16;
    const bf16* krl = Klo + (size_t)mykey * D_DIM + hl16;
    v8f s = vzero8();
    for (int kc = 0; kc < 32; ++kc) {
      const bf16* qrow = qh_s + ln15 * D_DIM + kc * 32;
      const bf16* qrlo = ql_s + ln15 * D_DIM + kc * 32;
      v16bf ah = load_a_frag(qrow, hl8);
      v16bf al = load_a_frag(qrlo, hl8);
      v16bf bh = load_b_frag(krh + kc * 32);
      v16bf bl = load_b_frag(krl + kc * 32);
      s = wmma_bf16(ah, bh, s);
      s = wmma_bf16(ah, bl, s);
      s = wmma_bf16(al, bh, s);
    }
#pragma unroll
    for (int i = 0; i < 8; ++i)
      Sb[(i + hl8) * 128 + wv * 16 + ln15] = s[i];
    __syncthreads();

    // ---- online softmax over the 128-key block (16 rows x 128 cols) ----
    const int r = tid >> 4, seg = tid & 15;
    {  // per-thread max of 8 scores
      const float* sr = Sb + r * 128 + seg * 8;
      float lm = sr[0];
#pragma unroll
      for (int j = 1; j < 8; ++j) lm = fmaxf(lm, sr[j]);
      red[r * 16 + seg] = lm;
    }
    __syncthreads();
    if (tid < 16) {
      float bm = red[tid * 16];
#pragma unroll
      for (int j = 1; j < 16; ++j) bm = fmaxf(bm, red[tid * 16 + j]);
      float mo = mrow[tid];
      float mn = fmaxf(mo, bm);
      crow[tid] = __expf(mo - mn);
      mrow[tid] = mn;
    }
    __syncthreads();
    {  // exponentiate, store bf16 probs, partial sums
      float mn = mrow[r];
      const float* sr = Sb + r * 128 + seg * 8;
      bf16* pr = Pb + r * 128 + seg * 8;
      float ls = 0.0f;
#pragma unroll
      for (int j = 0; j < 8; ++j) {
        float p = __expf(sr[j] - mn);
        pr[j] = to_bf(p);
        ls += p;
      }
      red[r * 16 + seg] = ls;
    }
    __syncthreads();
    if (tid < 16) {
      float ssum = 0.0f;
#pragma unroll
      for (int j = 0; j < 16; ++j) ssum += red[tid * 16 + j];
      lrow[tid] = lrow[tid] * crow[tid] + ssum;
    }
    __syncthreads();

    // ---- rescale O and accumulate O += P * V ----
    float cv[8];
#pragma unroll
    for (int i = 0; i < 8; ++i) cv[i] = crow[i + hl8];
#pragma unroll
    for (int nt = 0; nt < 8; ++nt)
#pragma unroll
      for (int i = 0; i < 8; ++i) acc[nt][i] *= cv[i];

    for (int c = 0; c < 4; ++c) {               // 128 keys = 4 x K32
      v16bf pa = load_a_frag(Pb + ln15 * 128 + c * 32, hl8);
      const int keyc = keyblock + c * 32 + hl16;
      if (c == 0)  // pull next key block of K toward L2 while we do P*V
        __builtin_prefetch(Khi + (size_t)(mykey + 128) * D_DIM, 0, 1);
#pragma unroll
      for (int nt = 0; nt < 8; ++nt) {
        const bf16* vp =
            Vt + (size_t)(wv * 128 + nt * 16 + ln15) * R_TOT + keyc;
        v16bf vb = load_b_frag(vp);
        acc[nt] = wmma_bf16(pa, vb, acc[nt]);
      }
    }
    __syncthreads();  // protect Sb/Pb/crow before next iteration
  }

  // ---- finalize: O /= l, residual h = x + O, LayerNorm over D ----
  if (tid < 16) { sumr[tid] = 0.0f; sumsq[tid] = 0.0f; }
  __syncthreads();

  float lv[8];
#pragma unroll
  for (int i = 0; i < 8; ++i) lv[i] = 1.0f / lrow[i + hl8];

  float rs[8], rq[8];
#pragma unroll
  for (int i = 0; i < 8; ++i) { rs[i] = 0.0f; rq[i] = 0.0f; }

#pragma unroll
  for (int nt = 0; nt < 8; ++nt) {
    const int col = wv * 128 + nt * 16 + ln15;
#pragma unroll
    for (int i = 0; i < 8; ++i) {
      const int rowg = qbase + i + hl8;
      float h = x[(size_t)rowg * D_DIM + col] + acc[nt][i] * lv[i];
      acc[nt][i] = h;  // keep h in registers for the write-back pass
      rs[i] += h;
      rq[i] += h * h;
    }
  }
#pragma unroll
  for (int i = 0; i < 8; ++i) {
    atomicAdd(&sumr[i + hl8], rs[i]);
    atomicAdd(&sumsq[i + hl8], rq[i]);
  }
  __syncthreads();
  if (tid < 16) {
    float mu  = sumr[tid] * (1.0f / D_DIM);
    float var = sumsq[tid] * (1.0f / D_DIM) - mu * mu;
    mrow[tid] = mu;                       // reuse LDS for LN stats
    crow[tid] = rsqrtf(var + LN_EPS);
  }
  __syncthreads();

  float mu_i[8], rstd_i[8];
#pragma unroll
  for (int i = 0; i < 8; ++i) {
    mu_i[i]   = mrow[i + hl8];
    rstd_i[i] = crow[i + hl8];
  }
#pragma unroll
  for (int nt = 0; nt < 8; ++nt) {
    const int col = wv * 128 + nt * 16 + ln15;
    const float g  = gamma[col];
    const float be = beta[col];
#pragma unroll
    for (int i = 0; i < 8; ++i) {
      const int rowg = qbase + i + hl8;
      out[(size_t)rowg * D_DIM + col] =
          g * (acc[nt][i] - mu_i[i]) * rstd_i[i] + be;
    }
  }
}

// ---------------------------------------------------------------------------
// Host-side launcher.
// Workspace layout (bf16): Qhi | Qlo | Khi | Klo | Vt  = 5 x 16 MB = 80 MB.
// ---------------------------------------------------------------------------
extern "C" void kernel_launch(void* const* d_in, const int* in_sizes, int n_in,
                              void* d_out, int out_size, void* d_ws,
                              size_t ws_size, hipStream_t stream) {
  (void)in_sizes; (void)n_in; (void)out_size; (void)ws_size;
  const float* x     = (const float*)d_in[0];
  const float* Wq    = (const float*)d_in[1];
  const float* bq    = (const float*)d_in[2];
  const float* Wk    = (const float*)d_in[3];
  const float* bk    = (const float*)d_in[4];
  const float* Wv    = (const float*)d_in[5];
  const float* bv    = (const float*)d_in[6];
  const float* gamma = (const float*)d_in[7];
  const float* beta  = (const float*)d_in[8];

  const size_t N = (size_t)R_TOT * D_DIM;  // 8192*1024 elements
  bf16* Qhi = (bf16*)d_ws;
  bf16* Qlo = Qhi + N;
  bf16* Khi = Qlo + N;
  bf16* Klo = Khi + N;
  bf16* Vt  = Klo + N;

  dim3 gemm_grid(R_TOT / 128, D_DIM / 128);  // 64 x 8
  qkv_tile_kernel<<<gemm_grid, 256, 0, stream>>>(x, Wq, bq, Qhi, Qlo, 0);
  qkv_tile_kernel<<<gemm_grid, 256, 0, stream>>>(x, Wk, bk, Khi, Klo, 1);
  qkv_tile_kernel<<<gemm_grid, 256, 0, stream>>>(x, Wv, bv, Vt, nullptr, 2);

  // dynamic LDS: Qhi/Qlo tiles + score/prob buffers + reductions
  const size_t lds_bytes = 16 * D_DIM * 2 * 2   // qh_s + ql_s
                         + 16 * 128 * 4         // Sb
                         + 16 * 128 * 2         // Pb
                         + (16 * 3 + 256 + 32) * 4;  // m/l/c + red + sums
  attn_kernel<<<R_TOT / 16, 256, lds_bytes, stream>>>(
      x, Qhi, Qlo, Khi, Klo, Vt, gamma, beta, (float*)d_out);
}
